// Encoder_45741401703106
// MI455X (gfx1250) — matrix operand order repaired and checked by reading, not compiled, not run on hardware
//
#include <hip/hip_runtime.h>

// ---------------------------------------------------------------------------
// Problem constants (match reference)
// ---------------------------------------------------------------------------
static constexpr int BB = 4;        // batch
static constexpr int LL = 2048;     // sequence length
static constexpr int DM = 512;      // d_model
static constexpr int NH = 8;        // heads
static constexpr int HD = 64;       // head dim
static constexpr int DI = 2048;     // ffn inner
static constexpr int NL = 4;        // layers
static constexpr int MTOK = BB * LL; // 8192 token rows

typedef __bf16 bf16_t;
typedef __attribute__((ext_vector_type(8)))  __bf16 v8bf;
typedef __attribute__((ext_vector_type(16))) __bf16 v16bf;
typedef __attribute__((ext_vector_type(8)))  float  v8f;
typedef int v4i_t __attribute__((vector_size(4 * sizeof(int))));

#define WMMA_BF16(a, b, c) \
  __builtin_amdgcn_wmma_f32_16x16x32_bf16(false, (a), false, (b), (short)0, (c), false, false)

#if defined(__has_builtin)
#define HASB(x) __has_builtin(x)
#else
#define HASB(x) 0
#endif

#if HASB(__builtin_amdgcn_tensor_load_to_lds)
#define HAVE_TDM 1
#else
#define HAVE_TDM 0
#endif

// ---------------------------------------------------------------------------
// small device helpers
// ---------------------------------------------------------------------------
__device__ __forceinline__ bf16_t f2bf(float f) {
  union { float f; unsigned u; } in; in.f = f;
  unsigned u = in.u;
  unsigned r = (u + 0x7fffu + ((u >> 16) & 1u)) >> 16;   // round-nearest-even
  union { unsigned short s; bf16_t b; } out; out.s = (unsigned short)r;
  return out.b;
}

__device__ __forceinline__ v16bf ld16(const bf16_t* p) {  // 16 contiguous bf16
  v8bf lo = *(const v8bf*)p;
  v8bf hi = *(const v8bf*)(p + 8);
  v16bf r;
#pragma unroll
  for (int i = 0; i < 8; ++i) { r[i] = lo[i]; r[i + 8] = hi[i]; }
  return r;
}

__device__ __forceinline__ v16bf ld8x2(const bf16_t* p0, const bf16_t* p1) {
  v8bf lo = *(const v8bf*)p0;
  v8bf hi = *(const v8bf*)p1;
  v16bf r;
#pragma unroll
  for (int i = 0; i < 8; ++i) { r[i] = lo[i]; r[i + 8] = hi[i]; }
  return r;
}

__device__ __forceinline__ v8f vzero8() {
  v8f r;
#pragma unroll
  for (int i = 0; i < 8; ++i) r[i] = 0.f;
  return r;
}

// LDS byte address (offset within workgroup allocation) of a shared pointer
__device__ __forceinline__ unsigned lds_addr_of(const void* p) {
  return (unsigned)(size_t)(__attribute__((address_space(3))) const void*)p;
}

// 16B global -> LDS async copy (ASYNCcnt); GLOBAL_LOAD_ASYNC_TO_LDS_B128
__device__ __forceinline__ void async_copy16(const bf16_t* g, bf16_t* l) {
#if HASB(__builtin_amdgcn_global_load_async_to_lds_b128)
  __builtin_amdgcn_global_load_async_to_lds_b128(
      (__attribute__((address_space(1))) v4i_t*)(v4i_t*)g,
      (__attribute__((address_space(3))) v4i_t*)(v4i_t*)l,
      0, 0);
#else
  unsigned loff = lds_addr_of(l);
  asm volatile("global_load_async_to_lds_b128 %0, %1, off"
               :: "v"(loff), "v"(g) : "memory");
#endif
}

__device__ __forceinline__ void wait_async0() {
#if HASB(__builtin_amdgcn_s_wait_asynccnt)
  __builtin_amdgcn_s_wait_asynccnt(0);
#else
  asm volatile("s_wait_asynccnt 0" ::: "memory");
#endif
}

__device__ __forceinline__ void wait_tensor0() {
#if HASB(__builtin_amdgcn_s_wait_tensorcnt)
  __builtin_amdgcn_s_wait_tensorcnt(0);
#else
  asm volatile("s_wait_tensorcnt 0" ::: "memory");
#endif
}

#if HAVE_TDM
typedef __attribute__((ext_vector_type(4))) unsigned tdm_u32x4;
typedef __attribute__((ext_vector_type(8))) int      tdm_i32x8;
typedef __attribute__((ext_vector_type(4))) int      tdm_i32x4;

// TDM: DMA a 128-row x 32-elem bf16 tile (row stride = stride_elems) from
// global to LDS, padding the LDS destination by 4 DWORDs after every 16
// DWORDs (64B row -> 80B row stride). D# per ISA 8.3-8.5. TENSORcnt-tracked.
__device__ __forceinline__ void tdm_load_tile_128x32(const bf16_t* gsrc,
                                                     unsigned lds_byte_addr,
                                                     int stride_elems,
                                                     int total_rows) {
  unsigned long long ga = (unsigned long long)(size_t)gsrc;
  tdm_u32x4 g0;
  g0[0] = 1u;                                            // count=1, user mode
  g0[1] = lds_byte_addr;                                 // lds_addr
  g0[2] = (unsigned)(ga & 0xFFFFFFFFu);                  // global_addr lo
  g0[3] = (unsigned)((ga >> 32) & 0x1FFFFFFu) | (2u << 30); // addr hi | type=2
  tdm_i32x8 g1;
  g1[0] = (int)((1u << 16)      // data_size = 1 -> 2 bytes
              | (1u << 20)      // pad_enable
              | (3u << 22)      // pad_interval: 16 DWORDs
              | (3u << 25));    // pad_amount:   4 DWORDs
  g1[1] = (int)(((unsigned)stride_elems & 0xFFFFu) << 16);              // tensor_dim0[15:0]
  g1[2] = (int)((((unsigned)stride_elems >> 16) & 0xFFFFu)
              | (((unsigned)total_rows & 0xFFFFu) << 16));              // dim0 hi | dim1 lo
  g1[3] = (int)((((unsigned)total_rows >> 16) & 0xFFFFu) | (32u << 16)); // dim1 hi | tile_dim0=32
  g1[4] = (int)128u;                                                    // tile_dim1=128, tile_dim2=0
  g1[5] = (int)(unsigned)stride_elems;                                  // tensor_dim0_stride lo
  g1[6] = 0;
  g1[7] = 0;
  tdm_i32x4 gz = {0, 0, 0, 0};
#if defined(__clang_major__) && __clang_major__ >= 23
  tdm_i32x8 gz8 = {0, 0, 0, 0, 0, 0, 0, 0};
  __builtin_amdgcn_tensor_load_to_lds(g0, g1, gz, gz, gz8, 0);
#else
  __builtin_amdgcn_tensor_load_to_lds(g0, g1, gz, gz, 0);
#endif
}
#endif  // HAVE_TDM

// ---------------------------------------------------------------------------
// Weight prep: fp32 [K][N]  ->  bf16 [N][K]   (transposed, LDS-tiled)
// ---------------------------------------------------------------------------
__global__ __launch_bounds__(256) void k_transpose_cast(const float* __restrict__ src,
                                                        bf16_t* __restrict__ dst,
                                                        int K, int N) {
  __shared__ float tile[32][33];
  int tx = threadIdx.x & 31, ty = threadIdx.x >> 5;    // 32 x 8
  int kb = blockIdx.y * 32, nb = blockIdx.x * 32;
#pragma unroll
  for (int r = 0; r < 4; ++r)
    tile[ty + r * 8][tx] = src[(size_t)(kb + ty + r * 8) * N + nb + tx];
  __syncthreads();
#pragma unroll
  for (int r = 0; r < 4; ++r)
    dst[(size_t)(nb + ty + r * 8) * K + kb + tx] = f2bf(tile[tx][ty + r * 8]);
}

// ---------------------------------------------------------------------------
// Embedding lookup + temporal encoding (one block per token, D=512)
// ---------------------------------------------------------------------------
__global__ __launch_bounds__(256) void k_embed_tem(const int* __restrict__ etype,
                                                   const float* __restrict__ etime,
                                                   const float* __restrict__ npm,
                                                   const float* __restrict__ emb,
                                                   float* __restrict__ tem,
                                                   float* __restrict__ xf) {
  int t = blockIdx.x;
  float tm = etime[t], np = npm[t];
  int et = etype[t];
#pragma unroll
  for (int e = 0; e < 2; ++e) {
    int d = threadIdx.x + e * 256;
    // 10000^(-2*(d//2)/512) = exp(-ln(1e4)/512 * (d & ~1))
    float freq = __expf(-0.017988946039953095f * (float)(d & ~1));
    float r = tm * freq;
    float te = ((d & 1) == 0) ? __sinf(r) : __cosf(r);
    tem[(size_t)t * DM + d] = te * np;
    xf[(size_t)t * DM + d]  = emb[(size_t)et * DM + d];
  }
}

// x = x + tem; keep fp32 residual, emit bf16 copy
__global__ __launch_bounds__(256) void k_add_cast(float* __restrict__ xf,
                                                  const float* __restrict__ tem,
                                                  bf16_t* __restrict__ xb) {
  size_t i = (size_t)blockIdx.x * 256 + threadIdx.x;
  float s = xf[i] + tem[i];
  xf[i] = s;
  xb[i] = f2bf(s);
}

// v [B*L, 512] -> vt [(b,h,d)][L]   (per-head transposed V for P@V B-fragments)
__global__ __launch_bounds__(256) void k_transpose_v(const bf16_t* __restrict__ vb,
                                                     bf16_t* __restrict__ vt) {
  size_t i = (size_t)blockIdx.x * 256 + threadIdx.x;   // over B*L*512
  int c = (int)(i & 511);
  size_t t = i >> 9;
  int pos = (int)(t & (LL - 1));
  int b = (int)(t >> 11);
  int h = c >> 6, d = c & 63;
  vt[(((size_t)(b * NH + h) * HD + d) << 11) + pos] = vb[i];
}

// ---------------------------------------------------------------------------
// bf16 WMMA GEMM with double-buffered async LDS staging:
//   C[M,N] = act(A[M,K] @ BT[N,K]^T + bias) + resid
//   block = 256 thr (8 waves), block tile 128x128, wave tile 32x64.
//   A tile: per-lane GLOBAL_LOAD_ASYNC_TO_LDS_B128 (ASYNCcnt)
//   B tile: one TDM tensor_load_to_lds per k-step (TENSORcnt), LDS-padded
//   LDS rows padded 64B->80B (20-bank shift/row) for conflict-free ds_load_b128
//   Epilogue options are template params -> branch-free epilogues.
// ---------------------------------------------------------------------------
template <bool HAS_BIAS, bool HAS_RESID, bool GELU, bool OUT_F, bool OUT_B>
__global__ __launch_bounds__(256) void k_gemm_bf16(const bf16_t* __restrict__ A,
                                                   const bf16_t* __restrict__ BT,
                                                   const float* __restrict__ bias,
                                                   const float* __restrict__ resid,
                                                   float* __restrict__ Cf,
                                                   bf16_t* __restrict__ Cb,
                                                   int M, int N, int K) {
  __shared__ __align__(16) bf16_t ldsA[2][128 * 40];   // 40 bf16 = 80B row stride
  __shared__ __align__(16) bf16_t ldsB[2][128 * 40];

  const int tid  = threadIdx.x;
  const int lane = tid & 31;
  const int wave = tid >> 5;
  const int wm = wave & 3, wn = wave >> 2;
  const int gm = blockIdx.y * 128;
  const int gn = blockIdx.x * 128;
  const int lr = lane & 15, hi = lane >> 4;

  // per-thread 16B staging chunks: tile = 128 rows x 4 chunks = 512 chunks
  const int r0 = tid >> 2,          s0c = tid & 3;
  const int r1 = (tid + 256) >> 2,  s1c = tid & 3;      // (tid+256)&3 == tid&3

  auto stage = [&](int buf, int k0) {
    async_copy16(A + (size_t)(gm + r0) * K + k0 + s0c * 8, &ldsA[buf][r0 * 40 + s0c * 8]);
    async_copy16(A + (size_t)(gm + r1) * K + k0 + s1c * 8, &ldsA[buf][r1 * 40 + s1c * 8]);
#if HAVE_TDM
    if (wave == 0)
      tdm_load_tile_128x32(BT + (size_t)gn * K + k0, lds_addr_of(&ldsB[buf][0]), K, N);
#else
    async_copy16(BT + (size_t)(gn + r0) * K + k0 + s0c * 8, &ldsB[buf][r0 * 40 + s0c * 8]);
    async_copy16(BT + (size_t)(gn + r1) * K + k0 + s1c * 8, &ldsB[buf][r1 * 40 + s1c * 8]);
#endif
  };
  auto commit = [&]() {
    wait_async0();
#if HAVE_TDM
    if (wave == 0) wait_tensor0();
#endif
    __syncthreads();
  };

  v8f acc[2][4];
#pragma unroll
  for (int mi = 0; mi < 2; ++mi)
#pragma unroll
    for (int ni = 0; ni < 4; ++ni) acc[mi][ni] = vzero8();

  stage(0, 0);
  commit();

  int buf = 0;
  for (int k0 = 0; k0 < K; k0 += 32) {
    const bool more = (k0 + 32) < K;
    if (more) stage(buf ^ 1, k0 + 32);      // DMA next tile || compute this one

    v16bf af[2], bfr[4];
#pragma unroll
    for (int mi = 0; mi < 2; ++mi) {
      const int row = wm * 32 + mi * 16 + lr;
      af[mi] = ld8x2(&ldsA[buf][row * 40 + hi * 8],
                     &ldsA[buf][row * 40 + 16 + hi * 8]);
    }
#pragma unroll
    for (int ni = 0; ni < 4; ++ni) {
      const int row = wn * 64 + ni * 16 + lr;
      bfr[ni] = ld8x2(&ldsB[buf][row * 40 + hi * 8],
                      &ldsB[buf][row * 40 + 16 + hi * 8]);
    }
#pragma unroll
    for (int mi = 0; mi < 2; ++mi)
#pragma unroll
      for (int ni = 0; ni < 4; ++ni)
        acc[mi][ni] = WMMA_BF16(af[mi], bfr[ni], acc[mi][ni]);

    if (more) commit();
    buf ^= 1;
  }

  // epilogue: C layout -> lane holds col n = base+lr, rows i + hi*8
#pragma unroll
  for (int mi = 0; mi < 2; ++mi) {
#pragma unroll
    for (int ni = 0; ni < 4; ++ni) {
      const int n = gn + wn * 64 + ni * 16 + lr;
      float bs = 0.f;
      if (HAS_BIAS) bs = bias[n];
#pragma unroll
      for (int i = 0; i < 8; ++i) {
        const int m = gm + wm * 32 + mi * 16 + i + hi * 8;
        const size_t off = (size_t)m * N + n;
        float v = acc[mi][ni][i] + bs;
        if (HAS_RESID) v += resid[off];
        if (GELU) v = 0.5f * v * (1.0f + erff(v * 0.70710678118654752f)); // exact gelu
        if (OUT_F) Cf[off] = v;
        if (OUT_B) Cb[off] = f2bf(v);
      }
    }
  }
}

// ---------------------------------------------------------------------------
// Flash attention (causal + key-pad mask), bf16 WMMA, online softmax.
//   grid (L/64, NH, B), block 128 (4 waves); each wave owns a 16-query tile.
// ---------------------------------------------------------------------------
__global__ __launch_bounds__(128) void k_attention(const bf16_t* __restrict__ q,
                                                   const bf16_t* __restrict__ k,
                                                   const bf16_t* __restrict__ vt,
                                                   const int* __restrict__ etype,
                                                   bf16_t* __restrict__ out) {
  __shared__ bf16_t p_lds[4][16 * 32];

  const int wave = threadIdx.x >> 5;
  const int lane = threadIdx.x & 31;
  const int lr = lane & 15, hi = lane >> 4;
  const int b = blockIdx.z, h = blockIdx.y;
  const int q0 = blockIdx.x * 64 + wave * 16;
  const float scale = 0.125f;   // 1/sqrt(64)

  // preload Q tile (16 x 64) as two A-fragments
  const bf16_t* qrow = q + (size_t)(b * LL + q0 + lr) * DM + h * HD;
  v16bf qa[2];
#pragma unroll
  for (int kc = 0; kc < 2; ++kc) {
    const int base = kc * 32 + hi * 16;
    qa[kc] = ld16(qrow + base);
  }

  v8f o[4];
#pragma unroll
  for (int n = 0; n < 4; ++n) o[n] = vzero8();
  float mst[8], lst[8];
#pragma unroll
  for (int i = 0; i < 8; ++i) { mst[i] = -1e30f; lst[i] = 0.f; }

  const bf16_t* vbase = vt + ((size_t)(b * NH + h) * HD) * LL;  // [d][key]
  const int kend = q0 + 16;                                     // keys < kend

  for (int j0 = 0; j0 < kend; j0 += 32) {
    // ---- S = Q K^T for 32 keys (two 16-key column tiles) ----
    v8f s0 = vzero8(), s1 = vzero8();
    const bf16_t* krow0 = k + (size_t)(b * LL + j0 + lr) * DM + h * HD;
    const bf16_t* krow1 = krow0 + (size_t)16 * DM;
#pragma unroll
    for (int kc = 0; kc < 2; ++kc) {
      v16bf b0 = ld16(krow0 + kc * 32 + hi * 16);
      v16bf b1 = ld16(krow1 + kc * 32 + hi * 16);
      s0 = WMMA_BF16(qa[kc], b0, s0);
      s1 = WMMA_BF16(qa[kc], b1, s1);
    }

    // ---- mask + scale ----
    const int key0 = j0 + lr, key1 = j0 + 16 + lr;
    const bool pad0 = (etype[b * LL + key0] == 0);
    const bool pad1 = (etype[b * LL + key1] == 0);
#pragma unroll
    for (int i = 0; i < 8; ++i) {
      const int qi = q0 + i + hi * 8;
      float a0 = s0[i] * scale, a1 = s1[i] * scale;
      if (pad0 || key0 > qi) a0 = -1e9f;
      if (pad1 || key1 > qi) a1 = -1e9f;
      s0[i] = a0; s1[i] = a1;
    }

    // ---- online softmax (row reduce across 16-lane half) ----
#pragma unroll
    for (int i = 0; i < 8; ++i) {
      float rm = fmaxf(s0[i], s1[i]);
      rm = fmaxf(rm, __shfl_xor(rm, 1));
      rm = fmaxf(rm, __shfl_xor(rm, 2));
      rm = fmaxf(rm, __shfl_xor(rm, 4));
      rm = fmaxf(rm, __shfl_xor(rm, 8));
      const float mnew = fmaxf(mst[i], rm);
      const float corr = __expf(mst[i] - mnew);
      const float p0 = __expf(s0[i] - mnew);
      const float p1 = __expf(s1[i] - mnew);
      float rs = p0 + p1;
      rs += __shfl_xor(rs, 1);
      rs += __shfl_xor(rs, 2);
      rs += __shfl_xor(rs, 4);
      rs += __shfl_xor(rs, 8);
      lst[i] = lst[i] * corr + rs;
      mst[i] = mnew;
#pragma unroll
      for (int n = 0; n < 4; ++n) o[n][i] *= corr;
      const int row = i + hi * 8;
      p_lds[wave][row * 32 + lr]      = f2bf(p0);
      p_lds[wave][row * 32 + 16 + lr] = f2bf(p1);
    }

    // ---- P @ V (P 16x32 from LDS in A-fragment layout, V^T B-fragments) ----
    const bf16_t* pr = &p_lds[wave][lr * 32 + hi * 16];
    v16bf pa = ld16(pr);
#pragma unroll
    for (int n = 0; n < 4; ++n) {
      const int d = n * 16 + lr;
      v16bf bv = ld16(vbase + (size_t)d * LL + j0 + hi * 16);
      o[n] = WMMA_BF16(pa, bv, o[n]);
    }
  }

  // ---- normalize + store (out[token][h*64 + d], bf16) ----
#pragma unroll
  for (int i = 0; i < 8; ++i) {
    const float inv = 1.0f / lst[i];
    const int tok = b * LL + q0 + i + hi * 8;
#pragma unroll
    for (int n = 0; n < 4; ++n)
      out[(size_t)tok * DM + h * HD + n * 16 + lr] = f2bf(o[n][i] * inv);
  }
}

// ---------------------------------------------------------------------------
// LayerNorm * non_pad_mask; fp32 + bf16 outputs. One block per token.
// ---------------------------------------------------------------------------
__global__ __launch_bounds__(256) void k_layernorm(const float* __restrict__ in,
                                                   const float* __restrict__ g,
                                                   const float* __restrict__ be,
                                                   const float* __restrict__ npm,
                                                   float* __restrict__ outf,
                                                   bf16_t* __restrict__ outb) {
  __shared__ float sb[256];
  const int t = blockIdx.x, tid = threadIdx.x;
  const float* row = in + (size_t)t * DM;
  const float v0 = row[tid], v1 = row[tid + 256];

  sb[tid] = v0 + v1;
  __syncthreads();
  for (int off = 128; off > 0; off >>= 1) {
    if (tid < off) sb[tid] += sb[tid + off];
    __syncthreads();
  }
  const float mean = sb[0] * (1.0f / DM);
  __syncthreads();

  const float d0 = v0 - mean, d1 = v1 - mean;
  sb[tid] = d0 * d0 + d1 * d1;
  __syncthreads();
  for (int off = 128; off > 0; off >>= 1) {
    if (tid < off) sb[tid] += sb[tid + off];
    __syncthreads();
  }
  const float var = sb[0] * (1.0f / DM);
  const float rs = rsqrtf(var + 1e-6f);
  const float np = npm[t];

  const float o0 = (d0 * rs * g[tid] + be[tid]) * np;
  const float o1 = (d1 * rs * g[tid + 256] + be[tid + 256]) * np;
  outf[(size_t)t * DM + tid] = o0;
  outf[(size_t)t * DM + tid + 256] = o1;
  outb[(size_t)t * DM + tid] = f2bf(o0);
  outb[(size_t)t * DM + tid + 256] = f2bf(o1);
}

// ---------------------------------------------------------------------------
// Host orchestration
// ---------------------------------------------------------------------------
extern "C" void kernel_launch(void* const* d_in, const int* in_sizes, int n_in,
                              void* d_out, int out_size, void* d_ws, size_t ws_size,
                              hipStream_t stream) {
  (void)in_sizes; (void)n_in; (void)out_size; (void)ws_size;

  const int*   event_type = (const int*)d_in[0];
  const float* event_time = (const float*)d_in[1];
  const float* npm        = (const float*)d_in[2];
  const float* emb        = (const float*)d_in[3];
  const float* Wq         = (const float*)d_in[4];
  const float* Wk         = (const float*)d_in[5];
  const float* Wv         = (const float*)d_in[6];
  const float* fc_w       = (const float*)d_in[7];
  const float* fc_b       = (const float*)d_in[8];
  const float* ln1_g      = (const float*)d_in[9];
  const float* ln1_b      = (const float*)d_in[10];
  const float* W1         = (const float*)d_in[11];
  const float* b1         = (const float*)d_in[12];
  const float* W2         = (const float*)d_in[13];
  const float* b2         = (const float*)d_in[14];
  const float* ln2_g      = (const float*)d_in[15];
  const float* ln2_b      = (const float*)d_in[16];
  float* out = (float*)d_out;

  // ---- workspace carve-up (~126 MB) ----
  const size_t BLD = (size_t)MTOK * DM;        // 4,194,304 elems
  char* ws = (char*)d_ws;
  size_t off = 0;
  auto carve = [&](size_t bytes) {
    size_t o = off;
    off += (bytes + 255) & ~(size_t)255;
    return (void*)(ws + o);
  };
  float*  tem = (float*)carve(BLD * 4);
  float*  xf  = (float*)carve(BLD * 4);
  float*  yf  = (float*)carve(BLD * 4);
  bf16_t* xb  = (bf16_t*)carve(BLD * 2);
  bf16_t* ao  = (bf16_t*)carve(BLD * 2);
  bf16_t* qb  = (bf16_t*)carve(BLD * 2);
  bf16_t* kb  = (bf16_t*)carve(BLD * 2);
  bf16_t* vb  = (bf16_t*)carve(BLD * 2);
  bf16_t* vt  = (bf16_t*)carve(BLD * 2);
  bf16_t* hb  = qb;  // FFN hidden (B*L*2048 bf16) reuses dead q/k/v/vt region
  bf16_t* wT  = (bf16_t*)carve((size_t)NL * 3145728 * 2);

  const size_t WL = 3145728;  // bf16 weight elems per layer
  auto wqT = [&](int l) { return wT + (size_t)l * WL; };
  auto wkT = [&](int l) { return wT + (size_t)l * WL + 262144; };
  auto wvT = [&](int l) { return wT + (size_t)l * WL + 524288; };
  auto fcT = [&](int l) { return wT + (size_t)l * WL + 786432; };
  auto w1T = [&](int l) { return wT + (size_t)l * WL + 1048576; };
  auto w2T = [&](int l) { return wT + (size_t)l * WL + 2097152; };

  // ---- weight prep: fp32 -> transposed bf16 ----
  for (int l = 0; l < NL; ++l) {
    k_transpose_cast<<<dim3(16, 16), 256, 0, stream>>>(Wq + (size_t)l * DM * DM, wqT(l), DM, DM);
    k_transpose_cast<<<dim3(16, 16), 256, 0, stream>>>(Wk + (size_t)l * DM * DM, wkT(l), DM, DM);
    k_transpose_cast<<<dim3(16, 16), 256, 0, stream>>>(Wv + (size_t)l * DM * DM, wvT(l), DM, DM);
    k_transpose_cast<<<dim3(16, 16), 256, 0, stream>>>(fc_w + (size_t)l * DM * DM, fcT(l), DM, DM);
    k_transpose_cast<<<dim3(64, 16), 256, 0, stream>>>(W1 + (size_t)l * DM * DI, w1T(l), DM, DI);
    k_transpose_cast<<<dim3(16, 64), 256, 0, stream>>>(W2 + (size_t)l * DI * DM, w2T(l), DI, DM);
  }

  // ---- embedding + temporal encoding ----
  k_embed_tem<<<MTOK, 256, 0, stream>>>(event_type, event_time, npm, emb, tem, xf);

  const dim3 gN512(DM / 128, MTOK / 128);   // (4, 64)
  const dim3 gN2048(DI / 128, MTOK / 128);  // (16, 64)

  for (int l = 0; l < NL; ++l) {
    // x = x + tem  (xf = residual fp32, xb = bf16)
    k_add_cast<<<(int)(BLD / 256), 256, 0, stream>>>(xf, tem, xb);

    // Q, K, V projections (bf16 out only)
    k_gemm_bf16<false, false, false, false, true><<<gN512, 256, 0, stream>>>(
        xb, wqT(l), nullptr, nullptr, nullptr, qb, MTOK, DM, DM);
    k_gemm_bf16<false, false, false, false, true><<<gN512, 256, 0, stream>>>(
        xb, wkT(l), nullptr, nullptr, nullptr, kb, MTOK, DM, DM);
    k_gemm_bf16<false, false, false, false, true><<<gN512, 256, 0, stream>>>(
        xb, wvT(l), nullptr, nullptr, nullptr, vb, MTOK, DM, DM);
    k_transpose_v<<<(int)(BLD / 256), 256, 0, stream>>>(vb, vt);

    // attention
    k_attention<<<dim3(LL / 64, NH, BB), 128, 0, stream>>>(qb, kb, vt, event_type, ao);

    // out @ fc_w + fc_b + residual ; ln1 ; * mask   (bias+resid, fp32 out)
    k_gemm_bf16<true, true, false, true, false><<<gN512, 256, 0, stream>>>(
        ao, fcT(l), fc_b + (size_t)l * DM, xf, yf, nullptr, MTOK, DM, DM);
    k_layernorm<<<MTOK, 256, 0, stream>>>(yf, ln1_g + (size_t)l * DM, ln1_b + (size_t)l * DM, npm, xf, xb);

    // FFN: gelu(x @ W1 + b1) @ W2 + b2 + residual ; ln2 ; * mask
    k_gemm_bf16<true, false, true, false, true><<<gN2048, 256, 0, stream>>>(
        xb, w1T(l), b1 + (size_t)l * DI, nullptr, nullptr, hb, MTOK, DI, DM);
    k_gemm_bf16<true, true, false, true, false><<<gN512, 256, 0, stream>>>(
        hb, w2T(l), b2 + (size_t)l * DM, xf, yf, nullptr, MTOK, DM, DI);

    float* ln2_out = (l == NL - 1) ? out : xf;
    k_layernorm<<<MTOK, 256, 0, stream>>>(yf, ln2_g + (size_t)l * DM, ln2_b + (size_t)l * DM, npm, ln2_out, xb);
  }
}
